// SelfAttention_59897613910747
// MI455X (gfx1250) — matrix operand order repaired and checked by reading, not compiled, not run on hardware
//
#include <hip/hip_runtime.h>

#define BN   4
#define CC   128
#define CI   16
#define NN   4096
#define SPLIT 4

typedef __attribute__((ext_vector_type(16))) _Float16 v16h;
typedef __attribute__((ext_vector_type(8)))  _Float16 v8h;
typedef __attribute__((ext_vector_type(8)))  float    v8f;

// ---------------------------------------------------------------------------
// Kernel A: fused 1x1-conv QKV projection, f32 -> f16, attention-friendly
// layouts: qT,kT = [B,N,16] (row-contiguous per pixel), v = [B,16,N].
// ---------------------------------------------------------------------------
__global__ __launch_bounds__(256) void qkv_kernel(
    const float* __restrict__ x,
    const float* __restrict__ Wq, const float* __restrict__ bq,
    const float* __restrict__ Wk, const float* __restrict__ bk,
    const float* __restrict__ Wv, const float* __restrict__ bv,
    _Float16* __restrict__ qT, _Float16* __restrict__ kT,
    _Float16* __restrict__ vv)
{
  const int t = blockIdx.x * 256 + threadIdx.x;   // one thread per (b, pixel)
  const int b = t >> 12;
  const int n = t & (NN - 1);
  const float* xp = x + (size_t)b * CC * NN + n;

  float qa[CI], ka[CI], va[CI];
  #pragma unroll
  for (int o = 0; o < CI; ++o) { qa[o] = bq[o]; ka[o] = bk[o]; va[o] = bv[o]; }

  for (int c = 0; c < CC; ++c) {                  // weights are wave-uniform -> s_loads
    const float xval = xp[(size_t)c * NN];
    #pragma unroll
    for (int o = 0; o < CI; ++o) {
      qa[o] = fmaf(Wq[o * CC + c], xval, qa[o]);
      ka[o] = fmaf(Wk[o * CC + c], xval, ka[o]);
      va[o] = fmaf(Wv[o * CC + c], xval, va[o]);
    }
  }

  v8h q0, q1, k0, k1;
  #pragma unroll
  for (int o = 0; o < 8; ++o) {
    q0[o] = (_Float16)qa[o]; q1[o] = (_Float16)qa[8 + o];
    k0[o] = (_Float16)ka[o]; k1[o] = (_Float16)ka[8 + o];
  }
  _Float16* qp = qT + (size_t)t * CI;             // (b*N+n)*16
  _Float16* kp = kT + (size_t)t * CI;
  *(v8h*)qp = q0; *(v8h*)(qp + 8) = q1;
  *(v8h*)kp = k0; *(v8h*)(kp + 8) = k1;
  #pragma unroll
  for (int o = 0; o < CI; ++o)
    vv[((size_t)b * CI + o) * NN + n] = (_Float16)va[o];
}

// ---------------------------------------------------------------------------
// Kernel B1: per-column softmax stats (softmax is over i for each (b,j)).
// One wave per column; online max/sum per lane, butterfly combine.
// ---------------------------------------------------------------------------
__global__ __launch_bounds__(256) void colstats_kernel(
    const _Float16* __restrict__ qT, const _Float16* __restrict__ kT,
    float* __restrict__ cmax, float* __restrict__ cinv)
{
  const int lane = threadIdx.x & 31;
  const int wid  = threadIdx.x >> 5;
  const int col  = blockIdx.x * 8 + wid;          // B*N columns total
  const int b    = col >> 12;
  const int j    = col & (NN - 1);

  float kc[CI];
  const _Float16* kp = kT + ((size_t)b * NN + j) * CI;
  #pragma unroll
  for (int c = 0; c < CI; ++c) kc[c] = (float)kp[c];

  float m = -3.0e38f, s = 0.0f;
  for (int i = lane; i < NN; i += 32) {
    const _Float16* qp = qT + ((size_t)b * NN + i) * CI;
    float l = 0.0f;
    #pragma unroll
    for (int c = 0; c < CI; ++c) l = fmaf((float)qp[c], kc[c], l);
    const float mn = fmaxf(m, l);
    s = s * __expf(m - mn) + __expf(l - mn);
    m = mn;
  }
  #pragma unroll
  for (int off = 16; off >= 1; off >>= 1) {
    const float mo = __shfl_xor(m, off, 32);
    const float so = __shfl_xor(s, off, 32);
    const float mn = fmaxf(m, mo);
    s = s * __expf(m - mn) + so * __expf(mo - mn);
    m = mn;
  }
  if (lane == 0) {
    cmax[b * NN + j] = m;
    cinv[b * NN + j] = 1.0f / s;
  }
}

// ---------------------------------------------------------------------------
// Kernel B2: fused  logits(WMMA) -> exp -> attn store (non-temporal) ->
// V x P (WMMA). One wave owns a 16-column j-tile and a 1024-row i-range
// (SPLIT=4). attn is written exactly once and streamed past L2 (TH=NT) so
// the reused f16 tensors (~2MB) stay cache-resident.
// ---------------------------------------------------------------------------
__global__ __launch_bounds__(128) void attn_fused_kernel(
    const _Float16* __restrict__ qT,   // [B,N,16]
    const _Float16* __restrict__ kT,   // [B,N,16]
    const _Float16* __restrict__ v,    // [B,16,N]
    const float* __restrict__ cmax,    // [B,N]
    const float* __restrict__ cinv,    // [B,N]
    float* __restrict__ attn,          // [B,N,N]  (output #2)
    float* __restrict__ av_part)       // [SPLIT,B,16,N]
{
  __shared__ _Float16 plds_all[4][16][40];   // per-wave P tile, padded pitch
  const int tid  = threadIdx.x;
  const int wid  = tid >> 5;
  const int lane = tid & 31;
  const int m16  = lane & 15;
  const int half = lane >> 4;

  const int wg    = blockIdx.x * 4 + wid;    // 4096 waves
  const int split = wg & (SPLIT - 1);
  const int jt    = wg >> 2;                 // 0..1023
  const int b     = jt >> 8;
  const int jbase = (jt & 255) * 16;
  const int j     = jbase + m16;
  const int i0    = split * (NN / SPLIT);
  const int i1    = i0 + (NN / SPLIT);

  _Float16 (*plds)[40] = plds_all[wid];

  // B-fragment of K for the logit WMMA: lane = column j; elements 0..7 map to
  // contraction index c = 8*half + e, elements 8..15 (c >= 16) carry the zero pad.
  v16h kf = {};
  {
    const _Float16* kp = kT + ((size_t)b * NN + j) * CI + 8 * half;
    v8h k0 = *(const v8h*)(kp);
    #pragma unroll
    for (int e = 0; e < 8; ++e) { kf[e] = k0[e]; }
  }

  const float cm     = cmax[b * NN + j];
  const float cinv_j = cinv[b * NN + j];

  v8f acc = {};
  const _Float16* vrow = v + ((size_t)b * CI + m16) * NN;
  float* attn_b = attn + (size_t)b * NN * NN;

  for (int ibase = i0; ibase < i1; ibase += 32) {
    // ---- two 16x16 logit tiles: A = q rows (K=c, 16 real + 16 zero) ----
    #pragma unroll
    for (int t = 0; t < 2; ++t) {
      const int irow = ibase + t * 16 + m16;   // A fragment: lane = row i
      v8h q0 = *(const v8h*)(qT + ((size_t)b * NN + irow) * CI + 8 * half);
      v16h qa = {};                             // elems 8..15 = K>=16 -> zero
      #pragma unroll
      for (int e = 0; e < 8; ++e) qa[e] = q0[e];

      v8f lz = {};
      v8f l = __builtin_amdgcn_wmma_f32_16x16x32_f16(
                  false, qa, false, kf, (short)0, lz, false, false);

      // D layout: this lane holds rows i = t*16 + 8*half + r, column j.
      v8h ph;
      #pragma unroll
      for (int r = 0; r < 8; ++r) {
        const int iabs = ibase + t * 16 + 8 * half + r;
        const float p = __expf(l[r] - cm) * cinv_j;
        // streaming store: 268MB written once, never re-read -> TH_NT
        __builtin_nontemporal_store(p, &attn_b[(size_t)iabs * NN + j]);
        ph[r] = (_Float16)p;
      }
      *(v8h*)&plds[m16][t * 16 + 8 * half] = ph;   // P stored [j][i]
    }
    // same-wave DS ops are in-order; wait + compiler barrier before re-read
    asm volatile("s_wait_dscnt 0x0" ::: "memory");

    // B-fragment of P: lane = column j, elems 0..7 -> i = 8*half + e, elems 8..15 -> i = 16 + 8*half + e.
    v8h p0 = *(const v8h*)&plds[m16][8 * half];
    v8h p1 = *(const v8h*)&plds[m16][16 + 8 * half];
    v16h pb;
    #pragma unroll
    for (int e = 0; e < 8; ++e) { pb[e] = p0[e]; pb[8 + e] = p1[e]; }

    // A-fragment of V: lane = row c; elems 0..7 -> i = ibase+8*half+e,
    // elems 8..15 -> i = ibase+16+8*half+e.
    v8h v0 = *(const v8h*)(vrow + ibase + 8 * half);
    v8h v1 = *(const v8h*)(vrow + ibase + 16 + 8 * half);
    v16h va;
    #pragma unroll
    for (int e = 0; e < 8; ++e) { va[e] = v0[e]; va[8 + e] = v1[e]; }

    acc = __builtin_amdgcn_wmma_f32_16x16x32_f16(
              false, va, false, pb, (short)0, acc, false, false);
  }

  // av D layout: lane = column j, rows c = 8*half + r. Deterministic partials.
  float* avp = av_part + (size_t)split * (BN * CI * NN);
  #pragma unroll
  for (int r = 0; r < 8; ++r)
    avp[((size_t)b * CI + 8 * half + r) * NN + jbase + m16] = acc[r];
}

// ---------------------------------------------------------------------------
// Kernel C: out = gamma * (Wf @ av + bf) + x   (sums the SPLIT partials;
// av partials stay resident in L2).
// ---------------------------------------------------------------------------
__global__ __launch_bounds__(256) void out_kernel(
    const float* __restrict__ av_part, const float* __restrict__ Wf,
    const float* __restrict__ bf, const float* __restrict__ x,
    const float* __restrict__ gamma, float* __restrict__ out)
{
  const int n  = blockIdx.x * 256 + threadIdx.x;
  const int co = blockIdx.y;
  const int b  = blockIdx.z;

  float s = bf[co];
  #pragma unroll
  for (int c = 0; c < CI; ++c) {
    float a = 0.0f;
    #pragma unroll
    for (int p = 0; p < SPLIT; ++p)
      a += av_part[(size_t)p * (BN * CI * NN) + ((size_t)b * CI + c) * NN + n];
    s = fmaf(Wf[co * CI + c], a, s);
  }
  const size_t idx = ((size_t)b * CC + co) * NN + n;
  out[idx] = fmaf(gamma[0], s, x[idx]);
}

// ---------------------------------------------------------------------------
extern "C" void kernel_launch(void* const* d_in, const int* in_sizes, int n_in,
                              void* d_out, int out_size, void* d_ws, size_t ws_size,
                              hipStream_t stream) {
  (void)in_sizes; (void)n_in; (void)out_size; (void)ws_size;
  const float* x     = (const float*)d_in[0];
  const float* gamma = (const float*)d_in[1];
  const float* Wq    = (const float*)d_in[2];
  const float* bq    = (const float*)d_in[3];
  const float* Wk    = (const float*)d_in[4];
  const float* bk    = (const float*)d_in[5];
  const float* Wv    = (const float*)d_in[6];
  const float* bv    = (const float*)d_in[7];
  const float* Wf    = (const float*)d_in[8];
  const float* bf    = (const float*)d_in[9];

  float* out  = (float*)d_out;                       // [B,C,H,W]
  float* attn = out + (size_t)BN * CC * NN;          // [B,N,N]

  char* ws = (char*)d_ws;
  _Float16* qT   = (_Float16*)(ws);                  // 512 KB  [B,N,16] f16
  _Float16* kT   = (_Float16*)(ws + (512 << 10));    // 512 KB  [B,N,16] f16
  _Float16* vv   = (_Float16*)(ws + (1024 << 10));   // 512 KB  [B,16,N] f16
  float*    cmax = (float*)(ws + (1536 << 10));      // 64 KB   [B,N]
  float*    cinv = (float*)(ws + (1600 << 10));      // 64 KB   [B,N]
  float*    avp  = (float*)(ws + (1664 << 10));      // 4 MB    [SPLIT,B,16,N]

  qkv_kernel<<<BN * NN / 256, 256, 0, stream>>>(x, Wq, bq, Wk, bk, Wv, bv,
                                                qT, kT, vv);
  colstats_kernel<<<BN * NN / 8, 256, 0, stream>>>(qT, kT, cmax, cinv);
  attn_fused_kernel<<<BN * (NN / 16) * SPLIT / 4, 128, 0, stream>>>(
      qT, kT, vv, cmax, cinv, attn, avp);
  out_kernel<<<dim3(NN / 256, CC, BN), 256, 0, stream>>>(avp, Wf, bf, x,
                                                         gamma, out);
}
